// LocalCausalAttention_35459249996092
// MI455X (gfx1250) — matrix operand order repaired and checked by reading, not compiled, not run on hardware
//
#include <hip/hip_runtime.h>

typedef __attribute__((ext_vector_type(16))) _Float16 v16h;
typedef __attribute__((ext_vector_type(8)))  float    v8f;
typedef __attribute__((ext_vector_type(4)))  int      v4i;
typedef __attribute__((ext_vector_type(8)))  int      v8i;
typedef __attribute__((ext_vector_type(4)))  unsigned int v4u;

#define B_  2
#define S_  2048
#define D_  1024
#define H_  16
#define DH_ 64
#define WIN_ 128

#if __has_builtin(__builtin_amdgcn_tensor_load_to_lds) && \
    __has_builtin(__builtin_amdgcn_s_wait_tensorcnt)
#define USE_TDM 1
#else
#define USE_TDM 0
#endif

#if __has_builtin(__builtin_amdgcn_global_load_async_to_lds_b128) && \
    __has_builtin(__builtin_amdgcn_s_wait_asynccnt)
#define USE_ASYNC_LDS 1
#else
#define USE_ASYNC_LDS 0
#endif

#define ASG __attribute__((address_space(1)))
#define ASL __attribute__((address_space(3)))

__device__ __forceinline__ ASG v4i* glb_ptr(const void* p) {
  return (ASG v4i*)(unsigned long long)p;         // flat == global address
}
__device__ __forceinline__ ASL v4i* lds_ptr(void* p) {
  // flat LDS address low 32 bits == DS offset (ISA aperture rules)
  return (ASL v4i*)(unsigned int)(unsigned long long)p;
}

union FragAB {
  v16h v;
  unsigned int u32[8];
  _Float16 h[16];
};

#if USE_TDM
// Issue one TDM 2D tile load: `rows` rows of 32 f16 elems, row stride
// `rowstride` elems, into LDS with 8-half padding after each row (-> 40-half
// LDS row pitch). D# bit packing per CDNA5 ISA 8.3/8.4.
__device__ __forceinline__ void tdm_load_tile(const void* gsrc, void* ldst,
                                              int rows, int rowstride) {
  unsigned long long ga = (unsigned long long)gsrc;
  unsigned int la = (unsigned int)(unsigned long long)ldst;
  v4u g0;
  g0.x = 1u;                                        // count=1 (valid), user mode
  g0.y = la;                                        // lds_addr (bytes)
  g0.z = (unsigned int)ga;                          // global_addr[31:0]
  g0.w = (unsigned int)(ga >> 32) | (2u << 30);     // global_addr[56:32] | type=2
  const unsigned td = 1u << 20;                     // huge tensor dims (no clip)
  v8i g1;
  g1[0] = (1 << 16)        // data_size = 2 bytes
        | (1 << 20)        // pad_enable
        | (3 << 22)        // pad_interval: 16 DWORDs (one 64B row)
        | (3 << 25);       // pad_amount: 4 DWORDs (8 halves)
  g1[1] = (int)((td & 0xFFFFu) << 16);              // tensor_dim0[15:0]
  g1[2] = (int)((td >> 16) | ((td & 0xFFFFu) << 16)); // dim0 hi | dim1 lo
  g1[3] = (int)((td >> 16) | (32u << 16));          // dim1 hi | tile_dim0=32
  g1[4] = rows & 0xFFFF;                            // tile_dim1 | tile_dim2=0
  g1[5] = rowstride;                                // tensor_dim0_stride[31:0]
  g1[6] = 0;                                        // stride hi | dim1_stride lo
  g1[7] = 0;
  v4i z4 = {};
  v8i z8 = {};
  __builtin_amdgcn_tensor_load_to_lds(g0, g1, z4, z4, z8, 0);
}
#endif

// ---------------------------------------------------------------------------
// fp32 -> f16 cast, vectorized (n multiple of 4)
// ---------------------------------------------------------------------------
__global__ __launch_bounds__(256)
void cast_f32_f16(const float* __restrict__ src, _Float16* __restrict__ dst, int n4) {
  int i = blockIdx.x * blockDim.x + threadIdx.x;
  if (i < n4) {
    float4 v = ((const float4*)src)[i];
    union { unsigned long long u; _Float16 h[4]; } o;
    o.h[0] = (_Float16)v.x; o.h[1] = (_Float16)v.y;
    o.h[2] = (_Float16)v.z; o.h[3] = (_Float16)v.w;
    ((unsigned long long*)dst)[i] = o.u;
  }
}

// ---------------------------------------------------------------------------
// fp32 [K][N] -> f16 transposed [N][K], 32x32 LDS tile
// ---------------------------------------------------------------------------
__global__ __launch_bounds__(256)
void castT_f32_f16(const float* __restrict__ src, _Float16* __restrict__ dst,
                   int K, int N) {
  __shared__ float tile[32][33];
  int n0 = blockIdx.x * 32, k0 = blockIdx.y * 32;
  int tr = threadIdx.x >> 5, tc = threadIdx.x & 31;
  #pragma unroll
  for (int rr = tr; rr < 32; rr += 8)
    tile[rr][tc] = src[(size_t)(k0 + rr) * N + n0 + tc];
  __syncthreads();
  #pragma unroll
  for (int rr = tr; rr < 32; rr += 8)
    dst[(size_t)(n0 + rr) * K + k0 + tc] = (_Float16)tile[tc][rr];
}

// ---------------------------------------------------------------------------
// Tiled f16 GEMM, C = A * Bt^T. A:[M][K], Bt:[N][K] (both K-contiguous).
// Block tile 128x128, Kstep 32, 8 waves, each wave 32x64 (8 WMMAs/Kstep).
// Double-buffered LDS fed by the TDM (or async global->LDS b128 copies).
// mode 0: fp32 C row-major. mode 1: QKV scatter (q/k [B,H,S,DH], vT [B,H,DH,S]).
// ---------------------------------------------------------------------------
#define BM 128
#define BN 128
#define BK 32
#define LDP (BK + 8)

__global__ __launch_bounds__(256)
void gemm_f16_wmma(const _Float16* __restrict__ A, const _Float16* __restrict__ Bt,
                   float* __restrict__ Cout,
                   _Float16* __restrict__ qout, _Float16* __restrict__ kout,
                   _Float16* __restrict__ vTout,
                   int M, int N, int K, int mode) {
  __shared__ __align__(16) _Float16 As[2][BM][LDP];
  __shared__ __align__(16) _Float16 Bs[2][BN][LDP];

  const int tid  = threadIdx.x;
  const int lane = tid & 31;
  const int wave = tid >> 5;
  const int wm   = wave & 3;            // 32-row sub-tile
  const int wn   = wave >> 2;           // 64-col sub-tile
  const int n0   = blockIdx.x * BN;
  const int m0   = blockIdx.y * BM;
  const int hi   = lane >> 4;
  const int lm   = lane & 15;

  const int r0 = tid >> 2,  c0 = (tid & 3) * 8;          // chunk 0: rows 0..63
  const int r1 = r0 + 64;                                // chunk 1: rows 64..127

  v8f acc[2][4] = {};

  auto stage = [&](int k0c, int bufi) {
#if USE_TDM
    if (wave == 0) {
      tdm_load_tile(A  + (size_t)m0 * K + k0c, &As[bufi][0][0], BM, K);
      tdm_load_tile(Bt + (size_t)n0 * K + k0c, &Bs[bufi][0][0], BN, K);
    }
#else
    const _Float16* ga0 = A  + (size_t)(m0 + r0) * K + k0c + c0;
    const _Float16* ga1 = A  + (size_t)(m0 + r1) * K + k0c + c0;
    const _Float16* gb0 = Bt + (size_t)(n0 + r0) * K + k0c + c0;
    const _Float16* gb1 = Bt + (size_t)(n0 + r1) * K + k0c + c0;
    _Float16* la0 = &As[bufi][r0][c0];
    _Float16* la1 = &As[bufi][r1][c0];
    _Float16* lb0 = &Bs[bufi][r0][c0];
    _Float16* lb1 = &Bs[bufi][r1][c0];
#if USE_ASYNC_LDS
    __builtin_amdgcn_global_load_async_to_lds_b128(glb_ptr(ga0), lds_ptr(la0), 0, 0);
    __builtin_amdgcn_global_load_async_to_lds_b128(glb_ptr(ga1), lds_ptr(la1), 0, 0);
    __builtin_amdgcn_global_load_async_to_lds_b128(glb_ptr(gb0), lds_ptr(lb0), 0, 0);
    __builtin_amdgcn_global_load_async_to_lds_b128(glb_ptr(gb1), lds_ptr(lb1), 0, 0);
#else
    *(float4*)la0 = *(const float4*)ga0;
    *(float4*)la1 = *(const float4*)ga1;
    *(float4*)lb0 = *(const float4*)gb0;
    *(float4*)lb1 = *(const float4*)gb1;
#endif
#endif
    if (k0c + BK < K) {                 // L2 warm-up for the tile after this one
      __builtin_prefetch(A  + (size_t)(m0 + r0) * K + k0c + c0 + BK, 0, 1);
      __builtin_prefetch(Bt + (size_t)(n0 + r0) * K + k0c + c0 + BK, 0, 1);
    }
  };

  const int T = K / BK;
  stage(0, 0);

  for (int it = 0; it < T; ++it) {
    const int cur = it & 1;
    if (it + 1 < T) stage((it + 1) * BK, cur ^ 1);
#if USE_TDM
    if (wave == 0) {                    // TDM per-wave in-order: <=2 left => stage `it` done
      if (it + 1 < T) __builtin_amdgcn_s_wait_tensorcnt(2);
      else            __builtin_amdgcn_s_wait_tensorcnt(0);
    }
#elif USE_ASYNC_LDS
    if (it + 1 < T) __builtin_amdgcn_s_wait_asynccnt(4);   // stage `it` landed
    else            __builtin_amdgcn_s_wait_asynccnt(0);
#endif
    __syncthreads();

    // A fragments (16x32): k = (r/4)*16 + hi*8 + (r%4)*2
    FragAB af[2];
    #pragma unroll
    for (int s = 0; s < 2; ++s) {
      #pragma unroll
      for (int r = 0; r < 8; ++r) {
        int kb = ((r >> 2) << 4) + (hi << 3) + ((r & 3) << 1);
        af[s].u32[r] = *(const unsigned int*)(&As[cur][wm * 32 + s * 16 + lm][kb]);
      }
    }
    // 4 B fragments (32x16): lane n=lm, k = (r/4)*16 + hi*8 + (r%4)*2
    #pragma unroll
    for (int g = 0; g < 4; ++g) {
      FragAB bf;
      #pragma unroll
      for (int r = 0; r < 8; ++r) {
        int kk = ((r >> 2) << 4) + (hi << 3) + ((r & 3) << 1);
        bf.u32[r] = *(const unsigned int*)(&Bs[cur][wn * 64 + g * 16 + lm][kk]);
      }
      acc[0][g] = __builtin_amdgcn_wmma_f32_16x16x32_f16(
          false, af[0].v, false, bf.v, (short)0, acc[0][g], false, false);
      acc[1][g] = __builtin_amdgcn_wmma_f32_16x16x32_f16(
          false, af[1].v, false, bf.v, (short)0, acc[1][g], false, false);
    }
    __syncthreads();
  }

  // epilogue: C layout row = r + 8*hi, col = lane%16
  #pragma unroll
  for (int s = 0; s < 2; ++s) {
    #pragma unroll
    for (int g = 0; g < 4; ++g) {
      #pragma unroll
      for (int r = 0; r < 8; ++r) {
        int mg = m0 + wm * 32 + s * 16 + r + hi * 8;
        int ng = n0 + wn * 64 + g * 16 + lm;
        float val = acc[s][g][r];
        if (mode == 0) {
          Cout[(size_t)mg * N + ng] = val;
        } else {
          int which = ng >> 10, rem = ng & 1023;
          int h = rem >> 6, dh = rem & 63;
          int b = mg >> 11, sq = mg & 2047;
          _Float16 hv = (_Float16)val;
          if (which == 0)      qout [(((size_t)b * H_ + h) * S_ + sq) * DH_ + dh] = hv;
          else if (which == 1) kout [(((size_t)b * H_ + h) * S_ + sq) * DH_ + dh] = hv;
          else                 vTout[(((size_t)b * H_ + h) * DH_ + dh) * S_ + sq] = hv;
        }
      }
    }
  }
}

// ---------------------------------------------------------------------------
// Windowed causal flash attention. One wave per 16-query block.
// q,k: [B,H,S,DH] f16; vT: [B,H,DH,S] f16; out: [B,S,D] f16
// ---------------------------------------------------------------------------
__global__ __launch_bounds__(128)
void attn_wmma(const _Float16* __restrict__ q, const _Float16* __restrict__ k,
               const _Float16* __restrict__ vT, _Float16* __restrict__ out) {
  __shared__ __align__(16) _Float16 P_lds[4][16][36];

  const int lane = threadIdx.x & 31;
  const int wave = threadIdx.x >> 5;
  const int h = blockIdx.y, b = blockIdx.z;
  const int q0 = (blockIdx.x * 4 + wave) * 16;
  const int hi = lane >> 4, lm = lane & 15;
  const size_t bh = (size_t)b * H_ + h;
  const _Float16* qp = q  + (bh * S_ + q0) * DH_;
  const _Float16* kp = k  + bh * S_ * DH_;
  const _Float16* vp = vT + bh * (size_t)DH_ * S_;

  FragAB qa[2];
  #pragma unroll
  for (int d = 0; d < 2; ++d) {
    #pragma unroll
    for (int r = 0; r < 8; ++r) {
      int dh = d * 32 + ((r >> 2) << 4) + (hi << 3) + ((r & 3) << 1);
      qa[d].u32[r] = *(const unsigned int*)(qp + (size_t)lm * DH_ + dh);
    }
  }

  v8f o[4] = {};
  float mrun[8], lrun[8];
  #pragma unroll
  for (int r = 0; r < 8; ++r) { mrun[r] = -1e30f; lrun[r] = 0.0f; }

  const int irow0 = q0 + hi * 8;
  int jlo = q0 - (WIN_ - 1); if (jlo < 0) jlo = 0; jlo &= ~31;

  for (int jt = jlo; jt <= q0 + 15; jt += 32) {
    v8f sc[2] = {};
    #pragma unroll
    for (int tg = 0; tg < 2; ++tg) {
      #pragma unroll
      for (int d = 0; d < 2; ++d) {
        FragAB kb;
        #pragma unroll
        for (int r = 0; r < 8; ++r) {
          int dh = d * 32 + ((r >> 2) << 4) + (hi << 3) + ((r & 3) << 1);
          int j  = jt + tg * 16 + lm;
          kb.u32[r] = *(const unsigned int*)(kp + (size_t)j * DH_ + dh);
        }
        sc[tg] = __builtin_amdgcn_wmma_f32_16x16x32_f16(
            false, qa[d].v, false, kb.v, (short)0, sc[tg], false, false);
      }
    }

    const float scale = 0.125f;
    #pragma unroll
    for (int r = 0; r < 8; ++r) {
      int i  = irow0 + r;
      int j0 = jt + lm, j1 = jt + 16 + lm;
      bool a0 = (j0 <= i) && (j0 + WIN_ > i);
      bool a1 = (j1 <= i) && (j1 + WIN_ > i);
      float s0v = a0 ? sc[0][r] * scale : -1e30f;
      float s1v = a1 ? sc[1][r] * scale : -1e30f;
      float t = fmaxf(s0v, s1v);
      #pragma unroll
      for (int off = 8; off >= 1; off >>= 1) t = fmaxf(t, __shfl_xor(t, off, 32));
      float mn = fmaxf(mrun[r], t);
      float p0 = a0 ? __expf(s0v - mn) : 0.0f;
      float p1 = a1 ? __expf(s1v - mn) : 0.0f;
      float alpha = __expf(mrun[r] - mn);
      float rs = p0 + p1;
      #pragma unroll
      for (int off = 8; off >= 1; off >>= 1) rs += __shfl_xor(rs, off, 32);
      lrun[r] = lrun[r] * alpha + rs;
      mrun[r] = mn;
      #pragma unroll
      for (int g = 0; g < 4; ++g) o[g][r] *= alpha;
      P_lds[wave][r + hi * 8][lm]      = (_Float16)p0;
      P_lds[wave][r + hi * 8][16 + lm] = (_Float16)p1;
    }
#if __has_builtin(__builtin_amdgcn_s_wait_dscnt)
    __builtin_amdgcn_s_wait_dscnt(0);
#else
    asm volatile("s_wait_dscnt 0" ::: "memory");
#endif

    FragAB pa;
    #pragma unroll
    for (int r = 0; r < 8; ++r) {
      int kb2 = ((r >> 2) << 4) + (hi << 3) + ((r & 3) << 1);
      pa.u32[r] = *(const unsigned int*)(&P_lds[wave][lm][kb2]);
    }
    #pragma unroll
    for (int g = 0; g < 4; ++g) {
      FragAB vb;
      #pragma unroll
      for (int r = 0; r < 8; ++r) {
        int key = ((r >> 2) << 4) + (hi << 3) + ((r & 3) << 1);
        int dh  = g * 16 + lm;
        vb.u32[r] = *(const unsigned int*)(vp + (size_t)dh * S_ + jt + key);
      }
      o[g] = __builtin_amdgcn_wmma_f32_16x16x32_f16(
          false, pa.v, false, vb.v, (short)0, o[g], false, false);
    }
  }

  #pragma unroll
  for (int g = 0; g < 4; ++g) {
    #pragma unroll
    for (int r = 0; r < 8; ++r) {
      int srow = q0 + r + hi * 8;
      int dh   = g * 16 + lm;
      float val = o[g][r] / lrun[r];
      out[((size_t)b * S_ + srow) * D_ + h * DH_ + dh] = (_Float16)val;
    }
  }
}

// ---------------------------------------------------------------------------
extern "C" void kernel_launch(void* const* d_in, const int* in_sizes, int n_in,
                              void* d_out, int out_size, void* d_ws, size_t ws_size,
                              hipStream_t stream) {
  const float* x     = (const float*)d_in[0];
  const float* w_qkv = (const float*)d_in[1];
  const float* w_out = (const float*)d_in[2];
  float* out = (float*)d_out;

  _Float16* ws = (_Float16*)d_ws;
  size_t off = 0;
  _Float16* xh   = ws + off; off += (size_t)B_ * S_ * D_;
  _Float16* wqT  = ws + off; off += (size_t)D_ * 3 * D_;   // [3D][D] transposed
  _Float16* woT  = ws + off; off += (size_t)D_ * D_;       // [D][D] transposed
  _Float16* qb   = ws + off; off += (size_t)B_ * H_ * S_ * DH_;
  _Float16* kb   = ws + off; off += (size_t)B_ * H_ * S_ * DH_;
  _Float16* vTb  = ws + off; off += (size_t)B_ * H_ * S_ * DH_;
  _Float16* ah   = ws + off; off += (size_t)B_ * S_ * D_;
  (void)ws_size; (void)n_in; (void)in_sizes; (void)out_size;

  // 1) casts (+ weight transposes to [N][K])
  {
    int n4 = (B_ * S_ * D_) / 4;
    cast_f32_f16<<<(n4 + 255) / 256, 256, 0, stream>>>(x, xh, n4);
    dim3 g1((3 * D_) / 32, D_ / 32);
    castT_f32_f16<<<g1, 256, 0, stream>>>(w_qkv, wqT, D_, 3 * D_);
    dim3 g2(D_ / 32, D_ / 32);
    castT_f32_f16<<<g2, 256, 0, stream>>>(w_out, woT, D_, D_);
  }

  // 2) QKV projection -> q/k/vT (f16, per-head layouts)
  {
    dim3 grid((3 * D_) / BN, (B_ * S_) / BM);
    gemm_f16_wmma<<<grid, 256, 0, stream>>>(xh, wqT, nullptr, qb, kb, vTb,
                                            B_ * S_, 3 * D_, D_, 1);
  }

  // 3) windowed causal attention -> ah (f16 [B,S,D])
  {
    dim3 grid(S_ / 64, H_, B_);
    attn_wmma<<<grid, 128, 0, stream>>>(qb, kb, vTb, ah);
  }

  // 4) output projection -> fp32 d_out
  {
    dim3 grid(D_ / BN, (B_ * S_) / BM);
    gemm_f16_wmma<<<grid, 256, 0, stream>>>(ah, woT, out, nullptr, nullptr, nullptr,
                                            B_ * S_, D_, D_, 0);
  }
}
